// EMMA_38792144617759
// MI455X (gfx1250) — matrix operand order repaired and checked by reading, not compiled, not run on hardware
//
#include <hip/hip_runtime.h>
#include <hip/hip_bf16.h>
#include <cstdint>
#include <cstddef>

// ---------------- model constants ----------------
#define BB      32
#define LL      64
#define TSTEPS  32      // only the write phase (t<32) can change M => affects output
#define EMB     256
#define HID     512
#define MEMD    256
#define NVALS   1024
#define NSLOTS  4096
#define KTOP    16
#define DIN     1024    // HID + EMB + MEMD
#define HIDDEN  1024
#define DEQI    8

typedef __bf16 bf16_t;
typedef __attribute__((ext_vector_type(16))) __bf16 v16bf;
typedef __attribute__((ext_vector_type(8)))  __bf16 v8bf;
typedef __attribute__((ext_vector_type(8)))  float  v8f;

// ---------------- WMMA core ----------------
__device__ __forceinline__ v8f wmma_bf16(v16bf a, v16bf b, v8f c) {
  // D = A(16x32 bf16) x B(32x16 bf16) + C(16x16 f32)
  return __builtin_amdgcn_wmma_f32_16x16x32_bf16(false, a, false, b, (short)0, c,
                                                 false, false);
}

__device__ __forceinline__ float softplusf_(float x) {
  return fmaxf(x, 0.0f) + log1pf(expf(-fabsf(x)));
}

// load 8 consecutive f32 (32B aligned)
__device__ __forceinline__ void load8f(float out[8], const float* src) {
  float4 a = *reinterpret_cast<const float4*>(src);
  float4 b = *reinterpret_cast<const float4*>(src + 4);
  out[0]=a.x; out[1]=a.y; out[2]=a.z; out[3]=a.w;
  out[4]=b.x; out[5]=b.y; out[6]=b.z; out[7]=b.w;
}

// A sources: [z | x_t | 0]  (v_t == 0 for all simulated steps)
__device__ __forceinline__ void load8_zx(float out[8], int m, int k,
                                         const float* z, const float* xt) {
  if (k < HID)            load8f(out, z  + (size_t)m * HID + k);
  else if (k < HID + EMB) load8f(out, xt + (size_t)m * EMB + (k - HID));
  else {
#pragma unroll
    for (int j = 0; j < 8; ++j) out[j] = 0.0f;
  }
}

// A sources: [z | h]
__device__ __forceinline__ void load8_zh(float out[8], int m, int k,
                                         const float* z, const float* h) {
  if (k < HID) load8f(out, z + (size_t)m * HID + k);
  else         load8f(out, h + (size_t)m * HID + (k - HID));
}

__device__ __forceinline__ v16bf pack16(const float g0[8], const float g1[8]) {
  v16bf a;
#pragma unroll
  for (int j = 0; j < 8; ++j) { a[j] = (bf16_t)g0[j]; a[j + 8] = (bf16_t)g1[j]; }
  return a;
}

// A fragment directly from row-major bf16 (row stride = HIDDEN)
__device__ __forceinline__ v16bf afrag_bf16(const bf16_t* row, int kb, int half) {
  v8bf a0 = *reinterpret_cast<const v8bf*>(row + kb + half * 8);
  v8bf a1 = *reinterpret_cast<const v8bf*>(row + kb + half * 8 + 16);
  v16bf a;
#pragma unroll
  for (int j = 0; j < 8; ++j) { a[j] = a0[j]; a[j + 8] = a1[j]; }
  return a;
}

// B fragment: two 8-bf16 chunks (k = kb+half*8.., kb+half*8+16..) from transposed-weight row (col-major weight)
__device__ __forceinline__ v16bf bfrag(const bf16_t* row, int kb, int half) {
  v8bf b0 = *reinterpret_cast<const v8bf*>(row + kb + half * 8);
  v8bf b1 = *reinterpret_cast<const v8bf*>(row + kb + half * 8 + 16);
  v16bf b;
#pragma unroll
  for (int j = 0; j < 8; ++j) { b[j] = b0[j]; b[j + 8] = b1[j]; }
  return b;
}

// ---------------- prep kernels ----------------
__global__ void k_transpose(bf16_t* __restrict__ dst, const float* __restrict__ src,
                            int K, int N) {  // src[K][N] -> dst[N][K] bf16
  int i = blockIdx.x * 256 + threadIdx.x;
  if (i >= K * N) return;
  int n = i / K, k = i - n * K;
  dst[i] = (bf16_t)src[(size_t)k * N + n];
}

__global__ void k_stackT(bf16_t* __restrict__ dst, const float* __restrict__ A,
                         const float* __restrict__ B2, int N) {
  // [[A];[B2]] : K=1024 (512+512), dst[N][1024]
  int i = blockIdx.x * 256 + threadIdx.x;
  if (i >= N * 1024) return;
  int n = i >> 10, k = i & 1023;
  float v = (k < HID) ? A[(size_t)k * N + n] : B2[(size_t)(k - HID) * N + n];
  dst[i] = (bf16_t)v;
}

__global__ void k_vproto(bf16_t* __restrict__ VPT, const float* __restrict__ value_W) {
  __shared__ float red[256];
  int n = blockIdx.x, tid = threadIdx.x;
  float v = value_W[(size_t)n * MEMD + tid];
  red[tid] = v * v; __syncthreads();
  for (int s = 128; s > 0; s >>= 1) { if (tid < s) red[tid] += red[tid + s]; __syncthreads(); }
  float nrm = fmaxf(sqrtf(red[0]), 1e-12f);
  VPT[(size_t)n * MEMD + tid] = (bf16_t)(v / nrm);
}

__global__ void k_embed(float* __restrict__ xf, const int* __restrict__ tokens,
                        const float* __restrict__ embW) {
  int i = blockIdx.x * 256 + threadIdx.x;  // TSTEPS*BB*EMB
  int e = i & 255;
  int rb = i >> 8;
  int b = rb & 31, t = rb >> 5;
  int tok = tokens[b * LL + t];
  xf[i] = embW[(size_t)tok * EMB + e];
}

__global__ void k_topk(int* __restrict__ topi, float* __restrict__ wslot,
                       const float* __restrict__ key_W, const int* __restrict__ key_ids,
                       const float* __restrict__ slot_keys) {
  __shared__ float kv[MEMD];
  __shared__ float sc[NSLOTS];
  __shared__ float red[256];
  __shared__ int   redi[256];
  __shared__ float selv[KTOP];
  __shared__ int   seli[KTOP];
  int b = blockIdx.x, tid = threadIdx.x;
  float v = key_W[(size_t)key_ids[b] * MEMD + tid];
  red[tid] = v * v; __syncthreads();
  for (int s = 128; s > 0; s >>= 1) { if (tid < s) red[tid] += red[tid + s]; __syncthreads(); }
  float nrm = fmaxf(sqrtf(red[0]), 1e-12f);
  kv[tid] = v / nrm;
  __syncthreads();
  for (int s0 = tid; s0 < NSLOTS; s0 += 256) {
    const float4* sk = reinterpret_cast<const float4*>(slot_keys + (size_t)s0 * MEMD);
    float acc = 0.0f;
    for (int e = 0; e < MEMD / 4; ++e) {
      float4 q = sk[e];
      acc += kv[4*e+0]*q.x + kv[4*e+1]*q.y + kv[4*e+2]*q.z + kv[4*e+3]*q.w;
    }
    sc[s0] = acc;
  }
  __syncthreads();
  for (int r = 0; r < KTOP; ++r) {
    float bv = -3.0e38f; int bi = NSLOTS;
    for (int s0 = tid; s0 < NSLOTS; s0 += 256) {
      float x = sc[s0];
      if (x > bv || (x == bv && s0 < bi)) { bv = x; bi = s0; }
    }
    red[tid] = bv; redi[tid] = bi; __syncthreads();
    for (int s = 128; s > 0; s >>= 1) {
      if (tid < s) {
        if (red[tid+s] > red[tid] || (red[tid+s] == red[tid] && redi[tid+s] < redi[tid])) {
          red[tid] = red[tid+s]; redi[tid] = redi[tid+s];
        }
      }
      __syncthreads();
    }
    if (tid == 0) { selv[r] = red[0]; seli[r] = redi[0]; sc[redi[0]] = -3.0e38f; }
    __syncthreads();
  }
  if (tid == 0) {
    float mx = -3.0e38f;
    for (int r = 0; r < KTOP; ++r) mx = fmaxf(mx, selv[r]);
    float s = 0.0f, e[KTOP];
    for (int r = 0; r < KTOP; ++r) { e[r] = expf(selv[r] - mx); s += e[r]; }
    for (int r = 0; r < KTOP; ++r) {
      wslot[b * KTOP + r] = e[r] / s;
      topi[b * KTOP + r] = seli[r];
    }
  }
}

__global__ void k_init(float* __restrict__ z, float* __restrict__ h,
                       const float* __restrict__ h0, float* __restrict__ M,
                       float* __restrict__ out) {
  size_t i = (size_t)blockIdx.x * 256 + threadIdx.x;  // NSLOTS*MEMD
  M[i] = 0.0f;
  if (i < BB * HID)  { z[i] = 0.0f; h[i] = h0[i & (HID - 1)]; }
  if (i < BB * NVALS) out[i] = 0.0f;
}

// ---------------- WMMA GEMM kernels ----------------
// H = tanh([z|x|0] @ u_W1 + b1)  -> bf16 [32,1024];  128 tiles, 16 blocks x 8 waves
__global__ void k_gemm1(bf16_t* __restrict__ Hact, const float* __restrict__ z,
                        const float* __restrict__ xt, const bf16_t* __restrict__ W1T,
                        const float* __restrict__ b1) {
  const int wv = threadIdx.x >> 5, lane = threadIdx.x & 31;
  const int tile = blockIdx.x * 8 + wv;   // 2 x 64 tiles
  const int tm = tile >> 6, tn = tile & 63;
  const int half = lane >> 4, l15 = lane & 15;
  const int mrow = tm * 16 + l15, ncol = tn * 16 + l15;
  const bf16_t* brow = W1T + (size_t)ncol * DIN;
  v8f acc;
#pragma unroll
  for (int r = 0; r < 8; ++r) acc[r] = 0.0f;
  for (int kb = 0; kb < DIN; kb += 32) {
    float g0[8], g1[8];
    load8_zx(g0, mrow, kb + half * 8,      z, xt);
    load8_zx(g1, mrow, kb + half * 8 + 16, z, xt);
    v16bf a = pack16(g0, g1);
    v16bf b = bfrag(brow, kb, half);
    if (kb + 64 < DIN) __builtin_prefetch(brow + kb + 64, 0, 1);
    acc = wmma_bf16(a, b, acc);
  }
  const float bias = b1[ncol];
#pragma unroll
  for (int r = 0; r < 8; ++r) {
    int mg = tm * 16 + half * 8 + r;
    Hact[(size_t)mg * HIDDEN + ncol] = (bf16_t)tanhf(acc[r] + bias);
  }
}

// z += 0.5*(H @ u_W2 + b2)   (== (1-R)z + R(z + H@W2 + b2), R=0.5); 8 blocks x 8 waves
__global__ void k_gemm2(float* __restrict__ z, const bf16_t* __restrict__ Hact,
                        const bf16_t* __restrict__ W2T, const float* __restrict__ b2) {
  const int wv = threadIdx.x >> 5, lane = threadIdx.x & 31;
  const int tile = blockIdx.x * 8 + wv;   // 2 x 32 tiles
  const int tm = tile >> 5, tn = tile & 31;
  const int half = lane >> 4, l15 = lane & 15;
  const int mrow = tm * 16 + l15, ncol = tn * 16 + l15;
  const bf16_t* arow = Hact + (size_t)mrow * HIDDEN;
  const bf16_t* brow = W2T + (size_t)ncol * HIDDEN;
  v8f acc;
#pragma unroll
  for (int r = 0; r < 8; ++r) acc[r] = 0.0f;
  for (int kb = 0; kb < HIDDEN; kb += 32) {
    v16bf a = afrag_bf16(arow, kb, half);
    v16bf b = bfrag(brow, kb, half);
    acc = wmma_bf16(a, b, acc);
  }
  const float bias = b2[ncol];
#pragma unroll
  for (int r = 0; r < 8; ++r) {
    int mg = tm * 16 + half * 8 + r;
    z[(size_t)mg * HID + ncol] += 0.5f * (acc[r] + bias);
  }
}

// g = tanh([z|h]@Wg + lb);  tau = softplus([z|h]@Wt + lbt) + 1;  16 blocks x 8 waves
__global__ void k_liquid(float* __restrict__ g, float* __restrict__ tau,
                         const float* __restrict__ z, const float* __restrict__ h,
                         const bf16_t* __restrict__ WGT, const bf16_t* __restrict__ WTT,
                         const float* __restrict__ lb, const float* __restrict__ lbt) {
  const int wv = threadIdx.x >> 5, lane = threadIdx.x & 31;
  const int tile = blockIdx.x * 8 + wv;     // 128: [0,64) -> g, [64,128) -> tau
  const bool isTau = tile >= 64;
  const int tl = tile & 63;
  const int tm = tl >> 5, tn = tl & 31;
  const int half = lane >> 4, l15 = lane & 15;
  const int mrow = tm * 16 + l15, ncol = tn * 16 + l15;
  const bf16_t* brow = (isTau ? WTT : WGT) + (size_t)ncol * 1024;
  v8f acc;
#pragma unroll
  for (int r = 0; r < 8; ++r) acc[r] = 0.0f;
  for (int kb = 0; kb < 1024; kb += 32) {
    float g0[8], g1[8];
    load8_zh(g0, mrow, kb + half * 8,      z, h);
    load8_zh(g1, mrow, kb + half * 8 + 16, z, h);
    v16bf a = pack16(g0, g1);
    v16bf b = bfrag(brow, kb, half);
    acc = wmma_bf16(a, b, acc);
  }
  const float bias = (isTau ? lbt : lb)[ncol];
#pragma unroll
  for (int r = 0; r < 8; ++r) {
    int mg = tm * 16 + half * 8 + r;
    float x = acc[r] + bias;
    if (isTau) tau[(size_t)mg * HID + ncol] = softplusf_(x) + 1.0f;
    else       g[(size_t)mg * HID + ncol]   = tanhf(x);
  }
}

__global__ void k_hupd(float* __restrict__ h, const float* __restrict__ g,
                       const float* __restrict__ tau) {
  int i = blockIdx.x * 256 + threadIdx.x;
  if (i >= BB * HID) return;
  float hv = h[i];
  h[i] = hv + (g[i] - hv) / tau[i];
}

// vpraw = [z|h] @ z2v_W + z2v_b ; 4 blocks x 8 waves (32 tiles)
__global__ void k_z2v(float* __restrict__ vpraw, const float* __restrict__ z,
                      const float* __restrict__ h, const bf16_t* __restrict__ ZVT,
                      const float* __restrict__ zb) {
  const int wv = threadIdx.x >> 5, lane = threadIdx.x & 31;
  const int tile = blockIdx.x * 8 + wv;   // 2 x 16 tiles
  const int tm = tile >> 4, tn = tile & 15;
  const int half = lane >> 4, l15 = lane & 15;
  const int mrow = tm * 16 + l15, ncol = tn * 16 + l15;
  const bf16_t* brow = ZVT + (size_t)ncol * 1024;
  v8f acc;
#pragma unroll
  for (int r = 0; r < 8; ++r) acc[r] = 0.0f;
  for (int kb = 0; kb < 1024; kb += 32) {
    float g0[8], g1[8];
    load8_zh(g0, mrow, kb + half * 8,      z, h);
    load8_zh(g1, mrow, kb + half * 8 + 16, z, h);
    v16bf a = pack16(g0, g1);
    v16bf b = bfrag(brow, kb, half);
    acc = wmma_bf16(a, b, acc);
  }
  const float bias = zb[ncol];
#pragma unroll
  for (int r = 0; r < 8; ++r) {
    int mg = tm * 16 + half * 8 + r;
    vpraw[(size_t)mg * MEMD + ncol] = acc[r] + bias;
  }
}

__global__ void k_vnorm(float* __restrict__ vp, const float* __restrict__ vpraw) {
  __shared__ float red[256];
  int b = blockIdx.x, tid = threadIdx.x;
  float v = vpraw[b * MEMD + tid];
  red[tid] = v * v; __syncthreads();
  for (int s = 128; s > 0; s >>= 1) { if (tid < s) red[tid] += red[tid + s]; __syncthreads(); }
  float nrm = fmaxf(sqrtf(red[0]), 1e-12f);
  vp[b * MEMD + tid] = v / nrm;
}

__global__ void k_decay(float* __restrict__ M, const int* __restrict__ wpos, int t) {
  __shared__ int anyw;
  if (threadIdx.x == 0) {
    int a = 0;
    for (int b = 0; b < BB; ++b) a |= (wpos[b] == t) ? 1 : 0;
    anyw = a;
  }
  __syncthreads();
  if (!anyw) return;
  size_t i = (size_t)blockIdx.x * 256 + threadIdx.x;
  M[i] *= 0.997f;
}

__global__ void k_scatter(float* __restrict__ M, const int* __restrict__ wpos, int t,
                          const int* __restrict__ topi, const float* __restrict__ wslot,
                          const float* __restrict__ vp) {
  int b = blockIdx.x;
  if (wpos[b] != t) return;
  int d = threadIdx.x;
  float val = vp[b * MEMD + d];
  for (int k = 0; k < KTOP; ++k)
    atomicAdd(&M[(size_t)topi[b * KTOP + k] * MEMD + d], wslot[b * KTOP + k] * val);
}

__global__ void k_final(float* __restrict__ vmem, float* __restrict__ scale_out,
                        const float* __restrict__ M, const int* __restrict__ topi,
                        const float* __restrict__ wslot, const float* __restrict__ lsr) {
  __shared__ float red[256];
  int b = blockIdx.x, tid = threadIdx.x;
  float acc = 0.0f;
  for (int k = 0; k < KTOP; ++k)
    acc += wslot[b * KTOP + k] * M[(size_t)topi[b * KTOP + k] * MEMD + tid];
  red[tid] = acc * acc; __syncthreads();
  for (int s = 128; s > 0; s >>= 1) { if (tid < s) red[tid] += red[tid + s]; __syncthreads(); }
  float nrm = fmaxf(sqrtf(red[0]), 1e-12f);
  vmem[b * MEMD + tid] = acc / nrm;
  if (b == 0 && tid == 0) scale_out[0] = softplusf_(lsr[0]) + 1e-3f;
}

// out = scale * (vmem_norm @ V_proto^T) ; 16 blocks x 8 waves (128 tiles), K=256
__global__ void k_logits(float* __restrict__ out, const float* __restrict__ vmem,
                         const bf16_t* __restrict__ VPT, const float* __restrict__ scale_p) {
  const int wv = threadIdx.x >> 5, lane = threadIdx.x & 31;
  const int tile = blockIdx.x * 8 + wv;   // 2 x 64 tiles
  const int tm = tile >> 6, tn = tile & 63;
  const int half = lane >> 4, l15 = lane & 15;
  const int mrow = tm * 16 + l15, ncol = tn * 16 + l15;
  const bf16_t* brow = VPT + (size_t)ncol * MEMD;
  v8f acc;
#pragma unroll
  for (int r = 0; r < 8; ++r) acc[r] = 0.0f;
  for (int kb = 0; kb < MEMD; kb += 32) {
    float g0[8], g1[8];
    load8f(g0, vmem + (size_t)mrow * MEMD + kb + half * 8);
    load8f(g1, vmem + (size_t)mrow * MEMD + kb + half * 8 + 16);
    v16bf a = pack16(g0, g1);
    v16bf b = bfrag(brow, kb, half);
    acc = wmma_bf16(a, b, acc);
  }
  const float s = scale_p[0];
#pragma unroll
  for (int r = 0; r < 8; ++r) {
    int mg = tm * 16 + half * 8 + r;
    out[(size_t)mg * NVALS + ncol] = s * acc[r];
  }
}

// ---------------- workspace layout (bytes) ----------------
static constexpr size_t OFF_W1T   = 0;                          // 1024*1024*2
static constexpr size_t OFF_W2T   = OFF_W1T   + 2097152;        // 512*1024*2
static constexpr size_t OFF_WGT   = OFF_W2T   + 1048576;        // 512*1024*2
static constexpr size_t OFF_WTT   = OFF_WGT   + 1048576;        // 512*1024*2
static constexpr size_t OFF_ZVT   = OFF_WTT   + 1048576;        // 256*1024*2
static constexpr size_t OFF_VPT   = OFF_ZVT   + 524288;         // 1024*256*2
static constexpr size_t OFF_XF    = OFF_VPT   + 524288;         // 32*32*256*4
static constexpr size_t OFF_M     = OFF_XF    + 1048576;        // 4096*256*4
static constexpr size_t OFF_Z     = OFF_M     + 4194304;        // 32*512*4
static constexpr size_t OFF_H     = OFF_Z     + 65536;
static constexpr size_t OFF_G     = OFF_H     + 65536;
static constexpr size_t OFF_TAU   = OFF_G     + 65536;
static constexpr size_t OFF_HACT  = OFF_TAU   + 65536;          // 32*1024*2
static constexpr size_t OFF_VPRAW = OFF_HACT  + 65536;          // 32*256*4
static constexpr size_t OFF_VP    = OFF_VPRAW + 32768;
static constexpr size_t OFF_VMEM  = OFF_VP    + 32768;
static constexpr size_t OFF_TOPI  = OFF_VMEM  + 32768;          // 32*16*4
static constexpr size_t OFF_WSLOT = OFF_TOPI  + 2048;
static constexpr size_t OFF_SCALE = OFF_WSLOT + 2048;

extern "C" void kernel_launch(void* const* d_in, const int* in_sizes, int n_in,
                              void* d_out, int out_size, void* d_ws, size_t ws_size,
                              hipStream_t stream) {
  (void)in_sizes; (void)n_in; (void)out_size; (void)ws_size;
  const int*   tokens    = (const int*)d_in[0];
  const int*   key_ids   = (const int*)d_in[1];
  const int*   write_pos = (const int*)d_in[2];
  const float* embed_W   = (const float*)d_in[5];
  const float* key_W     = (const float*)d_in[6];
  const float* value_W   = (const float*)d_in[7];
  const float* u_W1      = (const float*)d_in[8];
  const float* u_b1      = (const float*)d_in[9];
  const float* u_W2      = (const float*)d_in[10];
  const float* u_b2      = (const float*)d_in[11];
  const float* l_Wi      = (const float*)d_in[12];
  const float* l_Wh      = (const float*)d_in[13];
  const float* l_b       = (const float*)d_in[14];
  const float* l_Wti     = (const float*)d_in[15];
  const float* l_Wth     = (const float*)d_in[16];
  const float* l_bt      = (const float*)d_in[17];
  const float* h0        = (const float*)d_in[18];
  const float* z2v_W     = (const float*)d_in[19];
  const float* z2v_b     = (const float*)d_in[20];
  const float* lsr       = (const float*)d_in[21];
  const float* slot_keys = (const float*)d_in[22];

  char* ws = (char*)d_ws;
  bf16_t* W1T   = (bf16_t*)(ws + OFF_W1T);
  bf16_t* W2T   = (bf16_t*)(ws + OFF_W2T);
  bf16_t* WGT   = (bf16_t*)(ws + OFF_WGT);
  bf16_t* WTT   = (bf16_t*)(ws + OFF_WTT);
  bf16_t* ZVT   = (bf16_t*)(ws + OFF_ZVT);
  bf16_t* VPT   = (bf16_t*)(ws + OFF_VPT);
  float*  xf    = (float*)(ws + OFF_XF);
  float*  M     = (float*)(ws + OFF_M);
  float*  z     = (float*)(ws + OFF_Z);
  float*  h     = (float*)(ws + OFF_H);
  float*  g     = (float*)(ws + OFF_G);
  float*  tau   = (float*)(ws + OFF_TAU);
  bf16_t* Hact  = (bf16_t*)(ws + OFF_HACT);
  float*  vpraw = (float*)(ws + OFF_VPRAW);
  float*  vp    = (float*)(ws + OFF_VP);
  float*  vmem  = (float*)(ws + OFF_VMEM);
  int*    topi  = (int*)(ws + OFF_TOPI);
  float*  wslot = (float*)(ws + OFF_WSLOT);
  float*  scale = (float*)(ws + OFF_SCALE);
  float*  out   = (float*)d_out;

  // ---- one-time prep ----
  k_transpose<<<4096, 256, 0, stream>>>(W1T, u_W1, 1024, 1024);
  k_transpose<<<2048, 256, 0, stream>>>(W2T, u_W2, 1024, 512);
  k_stackT  <<<2048, 256, 0, stream>>>(WGT, l_Wi,  l_Wh,  512);
  k_stackT  <<<2048, 256, 0, stream>>>(WTT, l_Wti, l_Wth, 512);
  k_transpose<<<1024, 256, 0, stream>>>(ZVT, z2v_W, 1024, 256);
  k_vproto  <<<1024, 256, 0, stream>>>(VPT, value_W);
  k_embed   <<<1024, 256, 0, stream>>>(xf, tokens, embed_W);
  k_topk    <<<32,   256, 0, stream>>>(topi, wslot, key_W, key_ids, slot_keys);
  k_init    <<<4096, 256, 0, stream>>>(z, h, h0, M, out);

  // ---- write-phase recurrence (t in [0,32); M is frozen afterwards) ----
  for (int t = 0; t < TSTEPS; ++t) {
    const float* xt = xf + (size_t)t * BB * EMB;
    for (int it = 0; it < DEQI; ++it) {
      k_gemm1<<<16, 256, 0, stream>>>(Hact, z, xt, W1T, u_b1);
      k_gemm2<<<8,  256, 0, stream>>>(z, Hact, W2T, u_b2);
    }
    k_liquid <<<16, 256, 0, stream>>>(g, tau, z, h, WGT, WTT, l_b, l_bt);
    k_hupd   <<<64, 256, 0, stream>>>(h, g, tau);
    k_z2v    <<<4,  256, 0, stream>>>(vpraw, z, h, ZVT, z2v_b);
    k_vnorm  <<<32, 256, 0, stream>>>(vp, vpraw);
    k_decay  <<<4096, 256, 0, stream>>>(M, write_pos, t);
    k_scatter<<<32, 256, 0, stream>>>(M, write_pos, t, topi, wslot, vp);
  }

  // ---- final read + logits (covers every row's query step, all >= 32) ----
  k_final <<<32, 256, 0, stream>>>(vmem, scale, M, topi, wslot, lsr);
  k_logits<<<16, 256, 0, stream>>>(out, vmem, VPT, scale);
}
